// LatentCircuit_63273458204741
// MI455X (gfx1250) — matrix operand order repaired and checked by reading, not compiled, not run on hardware
//
#include <hip/hip_runtime.h>
#include <hip/hip_bf16.h>

typedef __bf16 v16bf __attribute__((ext_vector_type(16)));
typedef float  v8f   __attribute__((ext_vector_type(8)));

#define DIM_N    128     // hidden n
#define DIM_EMB  1024    // latent N
#define DIM_IN   6
#define DIM_OUT  2
#define DIM_T    512
#define DIM_B    256
#define EMB_ELEMS (DIM_EMB * DIM_T * DIM_B)   // 134217728

// q^T extended with W_out rows (2) + zero padding to a full 16-row tile
#define QT_ROWS  (DIM_EMB + 16)              // 1040
#define NC_TILES (QT_ROWS / 16)              // 65 (tile 64 = readout tile)
#define ST_TILE_BYTES (16 * DIM_N * 2)       // 4096

#define ALPHA_C     0.1f
#define ONE_MALPHA  0.9f
#define NOISE_SCALE 0.13416407865f   // sqrt(2/ALPHA * 0.03^2)

union Frag { v16bf v; unsigned u[8]; __bf16 h[16]; };

static __device__ __forceinline__ __bf16 f2bf(float f) {
    unsigned b = __builtin_bit_cast(unsigned, f);
    b = (b + 0x7FFFu + ((b >> 16) & 1u)) >> 16;       // round-to-nearest-even
    unsigned short s = (unsigned short)b;
    return __builtin_bit_cast(__bf16, s);
}

// CDNA5 async global->LDS copy (ASYNCcnt-tracked), 16 bytes per lane
static __device__ __forceinline__ void async_b128_to_lds(unsigned lds_byte_off,
                                                         const void* gptr) {
    unsigned long long ga = (unsigned long long)(uintptr_t)gptr;
    asm volatile("global_load_async_to_lds_b128 %0, %1, off"
                 :: "v"(lds_byte_off), "v"(ga) : "memory");
}
static __device__ __forceinline__ void wait_async0() {
    asm volatile("s_wait_asynccnt 0x0" ::: "memory");
}

// ---------------------------------------------------------------------------
// k0: precondition weights.
//   qT_bf [1040][128] bf16: rows 0-1023 = q^T, 1024-1025 = W_out, rest zero
//   wrec_bf [128][128] bf16: W_rec row-major
// ---------------------------------------------------------------------------
__global__ __launch_bounds__(256) void prep_kernel(const float* __restrict__ q,
                                                   const float* __restrict__ W_rec,
                                                   const float* __restrict__ W_out,
                                                   __bf16* __restrict__ qT_bf,
                                                   __bf16* __restrict__ wrec_bf) {
    int idx = blockIdx.x * 256 + threadIdx.x;
    if (idx < QT_ROWS * DIM_N) {
        int col = idx / DIM_N, k = idx % DIM_N;
        float v;
        if (col < DIM_EMB)                v = q[k * DIM_EMB + col];     // qT[col][k]
        else if (col < DIM_EMB + DIM_OUT) v = W_out[(col - DIM_EMB) * DIM_N + k];
        else                              v = 0.0f;
        qT_bf[idx] = f2bf(v);
    }
    if (idx < DIM_N * DIM_N) {
        wrec_bf[idx] = f2bf(W_rec[idx]);
    }
}

// ---------------------------------------------------------------------------
// k1: recurrent scan. grid = 16 (one per 16-batch tile), block = 256 (8 waves).
// Wave w owns the 16x16 (batch x n) tile with n-cols [16w, 16w+16).
// x_{t+1} = 0.9 x_t + 0.1 (relu(x_t W_rec^T + (u_t+inoise_t) W_inp^T) + rn_t)
// rec-noise loads are software-pipelined one step ahead of the serial chain.
// ---------------------------------------------------------------------------
__global__ __launch_bounds__(256) void scan_kernel(const float* __restrict__ u,
                                                   const float* __restrict__ rec_noise,
                                                   const float* __restrict__ inp_noise,
                                                   const float* __restrict__ W_inp,
                                                   const __bf16* __restrict__ wrec_bf,
                                                   __bf16* __restrict__ states_bf) {
    __shared__ __bf16 wrec_lds[DIM_N * DIM_N];   // 32 KB, W_rec row-major
    __shared__ __bf16 xs[16 * DIM_N];            // current state tile, 4 KB
    __shared__ float  sinp[16 * 8];              // (u + input-noise) per step

    const unsigned tid  = threadIdx.x;
    const unsigned wave = tid >> 5;
    const unsigned lane = tid & 31u;
    const unsigned lq   = lane & 15u;            // lane % 16
    const unsigned hi   = lane >> 4;             // 0 | 1
    const int b0 = blockIdx.x * 16;
    const unsigned j = wave * 16 + lq;           // this lane's n-column

    // stage W_rec into LDS (32-bit packed copies)
    {
        const unsigned* src = (const unsigned*)wrec_bf;
        unsigned* dst = (unsigned*)wrec_lds;
        for (int i = tid; i < DIM_N * DIM_N / 2; i += 256) dst[i] = src[i];
    }
    // zero x tile + states[:,0,:]
    for (int i = tid; i < 16 * DIM_N; i += 256) {
        xs[i] = __bf16(0.0f);
        int m = i / DIM_N, k = i % DIM_N;
        states_bf[(size_t)(b0 + m) * DIM_T * DIM_N + k] = __bf16(0.0f);
    }
    __syncthreads();

    // Resident B fragments: B[k][j] = W_rec[j][k]  (lane: N=lq; K split by lane half)
    Frag bw[4];
    {
        const unsigned* w32 = (const unsigned*)wrec_lds;
        for (int c = 0; c < 4; ++c) {
            unsigned base = (j * DIM_N + c * 32 + hi * 16) >> 1;
#pragma unroll
            for (int i = 0; i < 8; ++i) bw[c].u[i] = w32[base + i];
        }
    }
    // W_inp row for this lane's column
    float wi[DIM_IN];
#pragma unroll
    for (int i = 0; i < DIM_IN; ++i) wi[i] = W_inp[j * DIM_IN + i];

    // per-lane rec-noise base pointers + first-step preload (pipelined)
    const float* rnp[8];
    float rncur[8], xr[8];
#pragma unroll
    for (int r = 0; r < 8; ++r) {
        int m = r + (int)hi * 8;
        rnp[r] = rec_noise + ((size_t)(b0 + m) * DIM_T) * DIM_N + j;
        rncur[r] = rnp[r][0];
        xr[r] = 0.0f;
    }

    const unsigned* x32 = (const unsigned*)xs;

    for (int t = 0; t < DIM_T - 1; ++t) {
        // cooperative input staging: s[m][i] = u[i][t][b0+m] + scale*inp_noise[b0+m][t][i]
        if (tid < 16 * DIM_IN) {
            int i = tid % DIM_IN, m = tid / DIM_IN;
            float uv = u[((size_t)i * DIM_T + t) * DIM_B + b0 + m];
            float nv = inp_noise[((size_t)(b0 + m) * DIM_T + t) * DIM_IN + i];
            sinp[m * 8 + i] = uv + NOISE_SCALE * nv;
        }
        // prefetch next step's recurrent noise into registers (off critical path)
        float rnnext[8];
        {
            const size_t tn = (size_t)((t < DIM_T - 2) ? t + 1 : t) * DIM_N;
#pragma unroll
            for (int r = 0; r < 8; ++r) rnnext[r] = rnp[r][tn];
        }
        __syncthreads();

        // acc = x_tile @ W_rec^T  (K = 128 via 4x bf16 WMMA)
        v8f acc = {};
#pragma unroll
        for (int c = 0; c < 4; ++c) {
            Frag a;                                   // A[m][k]: M=lq, K contiguous (16*hi + s)
            unsigned base = (lq * DIM_N + c * 32 + hi * 16) >> 1;
#pragma unroll
            for (int i = 0; i < 4; ++i) a.u[i] = x32[base + i];
#pragma unroll
            for (int i = 0; i < 4; ++i) a.u[4 + i] = x32[base + 4 + i];
            acc = __builtin_amdgcn_wmma_f32_16x16x32_bf16(false, a.v, false, bw[c].v,
                                                          (short)0, acc, false, false);
        }

        // epilogue: input proj + relu + leak + rec noise; publish new state
#pragma unroll
        for (int r = 0; r < 8; ++r) {
            int m = r + (int)hi * 8;                  // C layout: M = r (+8 for hi lanes)
            float proj = acc[r];
#pragma unroll
            for (int i = 0; i < DIM_IN; ++i) proj += sinp[m * 8 + i] * wi[i];
            float relu = fmaxf(proj, 0.0f);
            xr[r] = ONE_MALPHA * xr[r] + ALPHA_C * (relu + NOISE_SCALE * rncur[r]);
            rncur[r] = rnnext[r];
            __bf16 hv = f2bf(xr[r]);
            xs[m * DIM_N + j] = hv;
            states_bf[((size_t)(b0 + m) * DIM_T + (t + 1)) * DIM_N + j] = hv;
        }
        __syncthreads();
    }
}

// ---------------------------------------------------------------------------
// k2: embedding + readout. The whole bf16 q^T (+ W_out rows) lives in LDS
// (266 KB of the 320 KB/WGP). Each block walks (t, b-tile) tasks: the 16x128
// states tile streams in via global_load_async_to_lds_b128 (double-buffered,
// ASYNCcnt), then 8 waves sweep 65 N-tiles with bf16 WMMA. Tile 64 is the
// W_out readout tile (2 valid rows).
// ---------------------------------------------------------------------------
__global__ __launch_bounds__(256) void embed_kernel(const __bf16* __restrict__ states_bf,
                                                    const __bf16* __restrict__ qT_bf,
                                                    float* __restrict__ out) {
    extern __shared__ char smem[];
    __bf16* q_lds   = (__bf16*)smem;                     // 1040*128 bf16 = 266240 B
    char*   st_base = smem + (size_t)QT_ROWS * DIM_N * 2;  // 2 x 4 KB tile buffers

    const unsigned tid  = threadIdx.x;
    const unsigned wave = tid >> 5;
    const unsigned lane = tid & 31u;
    const unsigned lq   = lane & 15u;
    const unsigned hi   = lane >> 4;
    const int row = tid >> 4, cg = tid & 15;             // this thread's staged uint4

    // runtime-computed LDS byte offset of this thread's 16B slot in buffer 0
    const unsigned st_off0 = (unsigned)(uintptr_t)st_base + tid * 16u;

    // stage q^T + W_out (uint4 copies)
    {
        const uint4* src = (const uint4*)qT_bf;
        uint4* dst = (uint4*)q_lds;
        for (int i = tid; i < QT_ROWS * DIM_N * 2 / 16; i += 256) dst[i] = src[i];
    }

    const unsigned* q32 = (const unsigned*)q_lds;
    const int num_tasks = DIM_T * (DIM_B / 16);          // 8192

    // kick off the first tile before the q barrier
    int task0 = blockIdx.x;
    if (task0 < num_tasks) {
        int t = task0 >> 4, b0 = (task0 & 15) * 16;
        async_b128_to_lds(st_off0,
            &states_bf[((size_t)(b0 + row) * DIM_T + t) * DIM_N + cg * 8]);
    }
    __syncthreads();                                     // q_lds ready

    unsigned buf = 0;
    for (int task = task0; task < num_tasks; task += gridDim.x) {
        const int t  = task >> 4;
        const int b0 = (task & 15) * 16;

        wait_async0();                                   // my async writes done
        __syncthreads();                                 // everyone's writes visible

        const int ntask = task + gridDim.x;              // prefetch next tile
        if (ntask < num_tasks) {
            int nt = ntask >> 4, nb0 = (ntask & 15) * 16;
            async_b128_to_lds(st_off0 + (buf ^ 1u) * (unsigned)ST_TILE_BYTES,
                &states_bf[((size_t)(nb0 + row) * DIM_T + nt) * DIM_N + cg * 8]);
        }

        // B fragments: B[k][b] = states[b0+b][t][k]
        const unsigned* st32 = (const unsigned*)(st_base + buf * ST_TILE_BYTES);
        Frag bs[4];
#pragma unroll
        for (int c = 0; c < 4; ++c) {
            unsigned base = (lq * DIM_N + c * 32 + hi * 16) >> 1;
#pragma unroll
            for (int i = 0; i < 8; ++i) bs[c].u[i] = st32[base + i];
        }

        // sweep latent tiles (0..63) + the readout tile (64)
        for (int ncT = wave; ncT < NC_TILES; ncT += 8) {
            const int nc0 = ncT * 16;
            v8f acc = {};
#pragma unroll
            for (int c = 0; c < 4; ++c) {
                Frag a;                                 // A[m][k] = qT[nc0+m][k]
                unsigned base = ((unsigned)(nc0 + lq) * DIM_N + c * 32 + hi * 16) >> 1;
#pragma unroll
                for (int i = 0; i < 4; ++i) a.u[i] = q32[base + i];
#pragma unroll
                for (int i = 0; i < 4; ++i) a.u[4 + i] = q32[base + 4 + i];
                acc = __builtin_amdgcn_wmma_f32_16x16x32_bf16(false, a.v, false, bs[c].v,
                                                              (short)0, acc, false, false);
            }
            if (ncT < DIM_EMB / 16) {
#pragma unroll
                for (int r = 0; r < 8; ++r) {
                    int nc = nc0 + r + (int)hi * 8;
                    out[(size_t)nc * DIM_T * DIM_B + (size_t)t * DIM_B + b0 + lq] = acc[r];
                }
            } else if (hi == 0) {                        // readout tile: rows 0-1 = W_out
#pragma unroll
                for (int r = 0; r < DIM_OUT; ++r)
                    out[(size_t)EMB_ELEMS + (size_t)r * DIM_T * DIM_B
                        + (size_t)t * DIM_B + b0 + lq] = acc[r];
            }
        }
        buf ^= 1;
    }
}

// ---------------------------------------------------------------------------
extern "C" void kernel_launch(void* const* d_in, const int* in_sizes, int n_in,
                              void* d_out, int out_size, void* d_ws, size_t ws_size,
                              hipStream_t stream) {
    const float* u         = (const float*)d_in[0];
    const float* rec_noise = (const float*)d_in[1];
    const float* inp_noise = (const float*)d_in[2];
    const float* W_inp     = (const float*)d_in[3];
    const float* W_rec     = (const float*)d_in[4];
    const float* W_out     = (const float*)d_in[5];
    const float* q         = (const float*)d_in[6];
    float* out = (float*)d_out;

    char* ws = (char*)d_ws;
    __bf16* states_bf = (__bf16*)ws;                                   // 32 MB
    __bf16* qT_bf     = (__bf16*)(ws + (size_t)DIM_B * DIM_T * DIM_N * 2);
    __bf16* wrec_bf   = (__bf16*)(ws + (size_t)DIM_B * DIM_T * DIM_N * 2
                                      + (size_t)QT_ROWS * DIM_N * 2);

    prep_kernel<<<(QT_ROWS * DIM_N + 255) / 256, 256, 0, stream>>>(q, W_rec, W_out,
                                                                   qT_bf, wrec_bf);
    scan_kernel<<<DIM_B / 16, 256, 0, stream>>>(u, rec_noise, inp_noise, W_inp,
                                                wrec_bf, states_bf);
    const size_t smem = (size_t)QT_ROWS * DIM_N * 2 + 2 * ST_TILE_BYTES;  // 274432 B
    embed_kernel<<<256, 256, smem, stream>>>(states_bf, qT_bf, out);
}
